// Network_45543833206809
// MI455X (gfx1250) — matrix operand order repaired and checked by reading, not compiled, not run on hardware
//
#include <hip/hip_runtime.h>
#include <hip/hip_bf16.h>
#include <float.h>

// ---------------------------------------------------------------------------
// Problem constants (N=16384 hardcoded; window schedule is uniform 2048 blocks)
// ---------------------------------------------------------------------------
#define NPTS  16384
#define NTGT  14336   // targets: points 2048..16383
#define NNB   16      // LOCAL_REGION
#define C128  128

// f16 workspace layout (element offsets, _Float16 units)
#define WQ_OFF    0        // 5*128*128, stored transposed [l][n][k]
#define WK_OFF    81920
#define WV_OFF    163840
#define PW2_OFF   245760   // pos_w2 transposed [n][k]
#define WOUT_OFF  262144   // wout transposed [n][k]
#define F16_TOTAL 278528
#define IDX_BYTE_OFF (F16_TOTAL * 2)   // then 14336*16 ints of KNN indices
// total ws need = 557056 + 917504 = 1474560 bytes

typedef __attribute__((ext_vector_type(16))) _Float16 v16h;
typedef __attribute__((ext_vector_type(8)))  float    v8f;

union V16H { v16h v; uint4 q[2]; _Float16 h[16]; };
union H8   { uint4 q; _Float16 h[8]; };

// wave-local LDS ordering: each wave owns a private LDS slice, so a DS-counter
// wait (split counter on gfx1250) is all that's needed between write/read phases.
__device__ __forceinline__ void wave_lds_sync() {
  __builtin_amdgcn_wave_barrier();
  asm volatile("s_wait_dscnt 0" ::: "memory");
  __builtin_amdgcn_wave_barrier();
}

// ---------------------------------------------------------------------------
// Fragment helpers (wave32 WMMA 16x16x32 f16 layouts per CDNA5 ISA 7.12.2)
// A (16x32 f16): lane m=ln&15; halves 0..7 -> K = kbase + (hi?8:0) + h,
//                halves 8..15 -> K = kbase + 16 + (hi?8:0) + (h-8)
// B (32x16 f16): lane n=ln&15; same K mapping as A
// C/D (16x16 f32): vgpr v, lane: element [m = v + (hi?8:0)][n = ln&15]
// ---------------------------------------------------------------------------
__device__ __forceinline__ v16h afrag(const _Float16* buf, int ld, int kbase) {
  const int ln = threadIdx.x & 31;
  const int m  = ln & 15;
  const int ko = kbase + ((ln >> 4) << 3);
  V16H r;
  r.q[0] = *(const uint4*)(buf + m * ld + ko);
  r.q[1] = *(const uint4*)(buf + m * ld + ko + 16);
  return r.v;
}

__device__ __forceinline__ v16h bfrag(const _Float16* Wt, int ld, int nbase, int kbase) {
  const int ln = threadIdx.x & 31;
  const _Float16* p = Wt + (nbase + (ln & 15)) * ld + kbase + ((ln >> 4) << 3);
  V16H r;
  r.q[0] = *(const uint4*)p;
  r.q[1] = *(const uint4*)(p + 16);
  return r.v;
}

// A fragment of the 16x16 attention matrix, K padded 16->32 with zeros
__device__ __forceinline__ v16h afrag_attn(const _Float16* abuf) {
  const int ln = threadIdx.x & 31;
  const int m  = ln & 15;
  V16H r;
  uint4 z; z.x = z.y = z.z = z.w = 0u;
  if (ln < 16) {
    r.q[0] = *(const uint4*)(abuf + m * 16);
    r.q[1] = *(const uint4*)(abuf + m * 16 + 8);
  } else {
    r.q[0] = z; r.q[1] = z;
  }
  return r.v;
}

// B fragment of v from (c,j)-major vtbuf; K(j) padded 16->32 (hi half-wave = 0)
__device__ __forceinline__ v16h bfrag_vt(const _Float16* vt, int t) {
  const int ln = threadIdx.x & 31;
  V16H r;
  uint4 z; z.x = z.y = z.z = z.w = 0u;
  if (ln < 16) {
    const _Float16* p = vt + (t * 16 + ln) * 16;
    r.q[0] = *(const uint4*)p;
    r.q[1] = *(const uint4*)(p + 8);
  } else {
    r.q[0] = z; r.q[1] = z;
  }
  return r.v;
}

// store 8 C-layout f32 tiles (16x128) to row-major f16 LDS
__device__ __forceinline__ void store_rm(_Float16* buf, const v8f (&acc)[8]) {
  const int ln = threadIdx.x & 31;
  const int c0 = ln & 15;
  const int mo = (ln >> 4) << 3;
#pragma unroll
  for (int t = 0; t < 8; t++)
#pragma unroll
    for (int v = 0; v < 8; v++)
      buf[(v + mo) * 128 + t * 16 + c0] = (_Float16)acc[t][v];
}

// store v (16x128) to (c,j)-major f16 LDS: one ds_store_b128 per tile per lane
__device__ __forceinline__ void store_vt(_Float16* vt, const v8f (&acc)[8]) {
  const int ln = threadIdx.x & 31;
  const int c0 = ln & 15;
  const int mo = (ln >> 4) << 3;
#pragma unroll
  for (int t = 0; t < 8; t++) {
    H8 h;
#pragma unroll
    for (int v = 0; v < 8; v++) h.h[v] = (_Float16)acc[t][v];
    *(uint4*)(vt + (t * 16 + c0) * 16 + mo) = h.q;
  }
}

// 16x128 @ 128x128 projection: A from LDS row-major f16, B from transposed f16 global
__device__ __forceinline__ void proj128(const _Float16* A, const _Float16* Wt, v8f (&out)[8]) {
#pragma unroll
  for (int t = 0; t < 8; t++) {
    v8f c = {};
#pragma unroll
    for (int kt = 0; kt < 4; kt++) {
      v16h a = afrag(A, 128, kt * 32);
      v16h b = bfrag(Wt, 128, t * 16, kt * 32);
      c = __builtin_amdgcn_wmma_f32_16x16x32_f16(false, a, false, b, (short)0, c, false, false);
    }
    out[t] = c;
  }
}

// ---------------------------------------------------------------------------
// Kernel 0: zero the output accumulator
// ---------------------------------------------------------------------------
__global__ void zero_out_kernel(float* out, int n) {
  int i = blockIdx.x * blockDim.x + threadIdx.x;
  if (i < n) out[i] = 0.0f;
}

// ---------------------------------------------------------------------------
// Kernel 1: stage weights as transposed f16 ([n][k]) so B fragments are two
// contiguous 16B global loads per lane. ~1MB -> stays L2 resident.
// ---------------------------------------------------------------------------
__global__ void prep_kernel(const float* __restrict__ wq, const float* __restrict__ wk,
                            const float* __restrict__ wv, const float* __restrict__ pw2,
                            const float* __restrict__ wout, _Float16* __restrict__ ws) {
  int e = blockIdx.x * 256 + threadIdx.x;
  if (e >= F16_TOTAL) return;
  if (e < WK_OFF) {
    int l = e >> 14, r = e & 16383, n = r >> 7, k = r & 127;
    ws[e] = (_Float16)wq[l * 16384 + k * 128 + n];
  } else if (e < WV_OFF) {
    int e2 = e - WK_OFF;
    int l = e2 >> 14, r = e2 & 16383, n = r >> 7, k = r & 127;
    ws[e] = (_Float16)wk[l * 16384 + k * 128 + n];
  } else if (e < PW2_OFF) {
    int e2 = e - WV_OFF;
    int l = e2 >> 14, r = e2 & 16383, n = r >> 7, k = r & 127;
    ws[e] = (_Float16)wv[l * 16384 + k * 128 + n];
  } else if (e < WOUT_OFF) {
    int e2 = e - PW2_OFF, n = e2 >> 7, k = e2 & 127;
    ws[e] = (_Float16)pw2[k * 128 + n];
  } else {
    int e2 = e - WOUT_OFF, n = e2 >> 7, k = e2 & 127;
    ws[e] = (_Float16)wout[k * 128 + n];
  }
}

// ---------------------------------------------------------------------------
// Kernel 2: KNN-16. One thread per target; point array (256KB) is L2-resident,
// top-16 kept in registers with branchless cndmask replacement.
// ---------------------------------------------------------------------------
__global__ void knn_kernel(const float* __restrict__ bx, int* __restrict__ idxb) {
  int t = blockIdx.x * blockDim.x + threadIdx.x;
  if (t >= NTGT) return;
  int p   = 2048 + t;
  int cur = p & ~2047;        // window start = context length (uniform 2048 schedule)
  float4 tg = *(const float4*)(bx + p * 4);
  float bd[NNB]; int bi[NNB];
#pragma unroll
  for (int i = 0; i < NNB; i++) { bd[i] = FLT_MAX; bi[i] = 0; }
  float worst = FLT_MAX; int wslot = 0;
  for (int n = 0; n < cur; n++) {
    float4 cg = *(const float4*)(bx + n * 4);
    float dx = tg.x - cg.x, dy = tg.y - cg.y, dz = tg.z - cg.z;
    float d = dx * dx + dy * dy + dz * dz;
    if (d < worst) {
#pragma unroll
      for (int i = 0; i < NNB; i++)
        if (i == wslot) { bd[i] = d; bi[i] = n; }
      worst = bd[0]; wslot = 0;
#pragma unroll
      for (int i = 1; i < NNB; i++)
        if (bd[i] > worst) { worst = bd[i]; wslot = i; }
    }
  }
#pragma unroll
  for (int i = 0; i < NNB; i++) idxb[t * NNB + i] = bi[i];
}

// ---------------------------------------------------------------------------
// Kernel 3: per-target transformer. One wave per target point; 4 waves/block.
// f (16x128) lives in registers as 8 f32 C-tiles; all GEMMs via v_wmma f16.
// ---------------------------------------------------------------------------
__global__ void __launch_bounds__(128)
main_kernel(const float* __restrict__ bx, const int* __restrict__ idxb,
            const _Float16* __restrict__ wsh,
            const float* __restrict__ attrw, const float* __restrict__ attrb,
            const float* __restrict__ posw1, const float* __restrict__ posb1,
            const float* __restrict__ posb2, const float* __restrict__ bout,
            const float* __restrict__ msw1, const float* __restrict__ msb1,
            const float* __restrict__ msw2, const float* __restrict__ msb2,
            const float* __restrict__ msw3, const float* __restrict__ msb3,
            float* __restrict__ out) {
  constexpr int PW = 12800;                 // per-wave LDS slice
  __shared__ __align__(16) char smem[4 * PW];
  const int ln  = threadIdx.x & 31;
  const int wid = threadIdx.x >> 5;
  char* base = smem + wid * PW;
  _Float16* fbuf  = (_Float16*)(base);          // 16x128 f16 (A staging, reused for q)
  _Float16* kbuf  = (_Float16*)(base + 4096);   // k row-major (j,c)
  _Float16* vtbuf = (_Float16*)(base + 8192);   // v (c,j)-major
  _Float16* abuf  = (_Float16*)(base + 12288);  // 16x16 attention
  float* featbuf = (float*)(base + 12288);      // alias abuf (post-attention)
  float* h1buf   = (float*)(base + 4096);       // alias kbuf
  float* h2buf   = (float*)(base + 8192);       // alias vtbuf

  const int t  = blockIdx.x * 4 + wid;          // 0..14335, exact
  const int p  = 2048 + t;
  const int c0 = ln & 15;
  const int mo = (ln >> 4) << 3;

  // ---- gather 16 neighbors; lanes 0-15 and 16-31 both hold neighbor j=ln&15
  const int j = ln & 15;
  const int gi = idxb[t * NNB + j];
  float4 nb = *(const float4*)(bx + gi * 4);
  float gx = nb.x, gy = nb.y, gz = nb.z, ga = nb.w;
  float sx = gx, sy = gy, sz = gz;
#pragma unroll
  for (int off = 1; off < 16; off <<= 1) {
    sx += __shfl_xor(sx, off, 32);
    sy += __shfl_xor(sy, off, 32);
    sz += __shfl_xor(sz, off, 32);
  }
  gx -= sx * (1.0f / 16.0f); gy -= sy * (1.0f / 16.0f); gz -= sz * (1.0f / 16.0f);
  float nrm = sqrtf(gx * gx + gy * gy + gz * gz);
  float r = nrm;
#pragma unroll
  for (int off = 1; off < 16; off <<= 1) r = fmaxf(r, __shfl_xor(r, off, 32));
  r = fmaxf(r, 1e-8f);
  gx /= r; gy /= r; gz /= r;

  // ---- pos MLP layer 1 (K=3, scalar): write hidden (16x128) f16 to fbuf
  {
    const int cb = (ln >> 4) << 6;   // lanes 0-15 -> c 0..63, 16-31 -> 64..127
    for (int cc = 0; cc < 64; cc++) {
      int c = cb + cc;
      float h = posb1[c] + gx * posw1[c] + gy * posw1[128 + c] + gz * posw1[256 + c];
      fbuf[j * 128 + c] = (_Float16)fmaxf(h, 0.0f);
    }
  }
  wave_lds_sync();

  // ---- pos MLP layer 2 via WMMA, then f init (attr embed + pos)
  v8f facc[8];
  {
    v8f pacc[8];
    proj128(fbuf, wsh + PW2_OFF, pacc);
    float gam[8];
#pragma unroll
    for (int v = 0; v < 8; v++) gam[v] = __shfl(ga, mo + v, 32);
#pragma unroll
    for (int tt = 0; tt < 8; tt++) {
      int c = tt * 16 + c0;
      float aw = attrw[c], ab = attrb[c], pb = posb2[c];
      v8f f;
#pragma unroll
      for (int v = 0; v < 8; v++) f[v] = pacc[tt][v] + pb + ab + gam[v] * aw;
      facc[tt] = f;
    }
  }

  // ---- 5 attention layers
  for (int l = 0; l < 5; l++) {
    const _Float16* wqT = wsh + WQ_OFF + l * 16384;
    const _Float16* wkT = wsh + WK_OFF + l * 16384;
    const _Float16* wvT = wsh + WV_OFF + l * 16384;

    store_rm(fbuf, facc);                 // f16 copy of f for A fragments
    wave_lds_sync();

    v8f tmp[8];
    proj128(fbuf, wkT, tmp);  store_rm(kbuf, tmp);   // k -> (j,c)
    proj128(fbuf, wvT, tmp);  store_vt(vtbuf, tmp);  // v -> (c,j)
    proj128(fbuf, wqT, tmp);                         // q in registers
    wave_lds_sync();
    store_rm(fbuf, tmp);                             // overwrite fbuf with q
    wave_lds_sync();

    // scores s = q @ k^T : A from fbuf(q), B from kbuf (N=j, K=c)
    v8f s = {};
#pragma unroll
    for (int kt = 0; kt < 4; kt++) {
      v16h qa = afrag(fbuf, 128, kt * 32);
      v16h kb = bfrag(kbuf, 128, 0, kt * 32);
      s = __builtin_amdgcn_wmma_f32_16x16x32_f16(false, qa, false, kb, (short)0, s, false, false);
    }

    // softmax over j (across the 16-lane group holding one row)
    float aval[8];
#pragma unroll
    for (int v = 0; v < 8; v++) {
      float x = s[v] * 0.08838834764831845f;   // 1/sqrt(128)
      float mx = x;
#pragma unroll
      for (int off = 1; off < 16; off <<= 1) mx = fmaxf(mx, __shfl_xor(mx, off, 32));
      float e = __expf(x - mx);
      float sm = e;
#pragma unroll
      for (int off = 1; off < 16; off <<= 1) sm += __shfl_xor(sm, off, 32);
      aval[v] = e / sm;
    }
#pragma unroll
    for (int v = 0; v < 8; v++)
      abuf[(v + mo) * 16 + c0] = (_Float16)aval[v];
    wave_lds_sync();

    // f += a @ v  (residual folded into WMMA C operand)
    v16h aa = afrag_attn(abuf);
#pragma unroll
    for (int tt = 0; tt < 8; tt++) {
      v16h vb = bfrag_vt(vtbuf, tt);
      facc[tt] = __builtin_amdgcn_wmma_f32_16x16x32_f16(false, aa, false, vb, (short)0, facc[tt], false, false);
    }
  }

  // ---- output projection + max-pool over neighbors
  store_rm(fbuf, facc);
  wave_lds_sync();
  {
    v8f oacc[8];
    proj128(fbuf, wsh + WOUT_OFF, oacc);
#pragma unroll
    for (int tt = 0; tt < 8; tt++) {
      float mx = oacc[tt][0];
#pragma unroll
      for (int v = 1; v < 8; v++) mx = fmaxf(mx, oacc[tt][v]);
      mx = fmaxf(mx, __shfl_xor(mx, 16, 32));       // combine m halves
      mx += bout[tt * 16 + c0];
      if (ln < 16) featbuf[tt * 16 + c0] = mx;
    }
  }
  wave_lds_sync();

  // ---- small MLP 128->64->16->2 (VALU; trivially small)
  {
    float h1a = msb1[ln], h1b = msb1[ln + 32];
    for (int i = 0; i < 128; i++) {
      float fv = featbuf[i];
      h1a += fv * msw1[i * 64 + ln];
      h1b += fv * msw1[i * 64 + ln + 32];
    }
    h1buf[ln]      = fmaxf(h1a, 0.0f);
    h1buf[ln + 32] = fmaxf(h1b, 0.0f);
    wave_lds_sync();
    if (ln < 16) {
      float h2 = msb2[ln];
      for (int i = 0; i < 64; i++) h2 += h1buf[i] * msw2[i * 16 + ln];
      h2buf[ln] = fmaxf(h2, 0.0f);
    }
    wave_lds_sync();
    if (ln == 0) {
      float mu = msb3[0], ls = msb3[1];
      for (int i = 0; i < 16; i++) {
        float h = h2buf[i];
        mu += h * msw3[i * 2];
        ls += h * msw3[i * 2 + 1];
      }
      float sg  = __expf(ls);
      float inv = 1.0f / (sg * 1.4142135623730951f);
      float ta  = bx[p * 4 + 3];
      float pr  = 0.5f * (erff((ta + 0.5f - mu) * inv) - erff((ta - 0.5f - mu) * inv));
      float bits = -log2f(pr + 1e-10f);
      bits = fminf(fmaxf(bits, 0.0f), 50.0f);
      atomicAdd(out, bits);
    }
  }
}

// ---------------------------------------------------------------------------
extern "C" void kernel_launch(void* const* d_in, const int* in_sizes, int n_in,
                              void* d_out, int out_size, void* d_ws, size_t ws_size,
                              hipStream_t stream) {
  (void)in_sizes; (void)n_in; (void)ws_size;
  const float* bx    = (const float*)d_in[0];
  const float* attrw = (const float*)d_in[1];
  const float* attrb = (const float*)d_in[2];
  const float* posw1 = (const float*)d_in[3];
  const float* posb1 = (const float*)d_in[4];
  const float* posw2 = (const float*)d_in[5];
  const float* posb2 = (const float*)d_in[6];
  const float* wq    = (const float*)d_in[7];
  const float* wk    = (const float*)d_in[8];
  const float* wv    = (const float*)d_in[9];
  const float* wout  = (const float*)d_in[10];
  const float* boutp = (const float*)d_in[11];
  const float* msw1  = (const float*)d_in[12];
  const float* msb1  = (const float*)d_in[13];
  const float* msw2  = (const float*)d_in[14];
  const float* msb2  = (const float*)d_in[15];
  const float* msw3  = (const float*)d_in[16];
  const float* msb3  = (const float*)d_in[17];
  float* out = (float*)d_out;
  _Float16* wsh = (_Float16*)d_ws;
  int* idxb = (int*)((char*)d_ws + IDX_BYTE_OFF);

  zero_out_kernel<<<(out_size + 255) / 256, 256, 0, stream>>>(out, out_size);
  prep_kernel<<<(F16_TOTAL + 255) / 256, 256, 0, stream>>>(wq, wk, wv, posw2, wout, wsh);
  knn_kernel<<<NTGT / 256, 256, 0, stream>>>(bx, idxb);
  main_kernel<<<NTGT / 4, 128, 0, stream>>>(bx, idxb, wsh,
                                            attrw, attrb, posw1, posb1, posb2, boutp,
                                            msw1, msb1, msw2, msb2, msw3, msb3, out);
}
